// MultiHeadAttentionEinsum_86981677679069
// MI455X (gfx1250) — compile-verified, matrix operand order checked by reading, not run on hardware
//
#include <hip/hip_runtime.h>
#include <hip/hip_bf16.h>

typedef __attribute__((ext_vector_type(16))) _Float16 v16h;
typedef __attribute__((ext_vector_type(8)))  _Float16 v8h;
typedef __attribute__((ext_vector_type(8)))  float    v8f;
typedef __attribute__((ext_vector_type(4)))  int      v4i;

// Async LDS->global store path (gfx1250): guard so compile never regresses.
#if defined(__has_builtin)
#if __has_builtin(__builtin_amdgcn_global_store_async_from_lds_b128)
#define HAVE_ASYNC_STORE 1
#endif
#if __has_builtin(__builtin_amdgcn_s_wait_asynccnt)
#define HAVE_WAIT_ASYNC 1
#endif
#endif

// b128 async ops move one 4xDWORD vector per lane; pointers are AS-qualified.
typedef __attribute__((address_space(1))) v4i* g4i_t;   // global
typedef __attribute__((address_space(3))) v4i* l4i_t;   // LDS

// Problem dims (compile-time)
constexpr int Bb = 2, Ss = 2048, Dd = 1024, Nn = 16, Hh = 64;
constexpr int MROWS = Bb * Ss;          // 4096
constexpr int FTILES = Ss / 16;         // 128 query-row tiles per (b,n)

// CDNA5 16-bit fragment layout: lanes 0-15 hold K {0..7,16..23},
// lanes 16-31 hold K {8..15,24..31}. Inverse map for staging:
//   half(k) = (k>>3)&1 ;  pos(k) = (k&7) + 8*(k>>4)
// so a lane's 16 half-elements are CONTIGUOUS 32B when stored frag-major.

enum OutMode { OUT_F32_ROWMAJOR = 0, OUT_F16_BNSH = 1, OUT_F16_BNHS = 2 };

// ---------------------------------------------------------------------------
// GEMM  C = A[M,K] * B[K,Ncol]  (A: f32 or f16 row-major, B: f32 row-major)
// 64x64 C tile / block, 4 waves, k-step 64 (8 WMMA per barrier interval),
// fragment-major LDS tiles -> one v16h (2x ds_load_b128) per fragment.
// ---------------------------------------------------------------------------
template <typename AT, int MODE>
__global__ __launch_bounds__(128) void wmma_gemm(
    const AT* __restrict__ A, const float* __restrict__ Bm,
    void* __restrict__ Cv, int M, int K, int Ncol)
{
    __shared__ alignas(32) _Float16 Af[4][2][32][16];  // [row-subtile][kk][lane][elt]
    __shared__ alignas(32) _Float16 Bf[4][2][32][16];  // [col-subtile][kk][lane][elt]

    const int tid  = threadIdx.x;
    const int wave = tid >> 5;
    const int lane = tid & 31;
    const int m0 = blockIdx.y * 64;
    const int n0 = blockIdx.x * 64;

    v8f acc[4] = {};

    for (int k0 = 0; k0 < K; k0 += 64) {
        // stage A 64x64 -> fragment-major (coalesced read along c)
        for (int i = tid; i < 64 * 64; i += 128) {
            const int r = i >> 6, c = i & 63;
            const int kk = c >> 5, cc = c & 31;
            const int hf = (cc >> 3) & 1, pos = (cc & 7) + 8 * (cc >> 4);
            Af[r >> 4][kk][(r & 15) + 16 * hf][pos] =
                (_Float16)A[(size_t)(m0 + r) * K + k0 + c];
        }
        // stage B 64x64 -> fragment-major
        for (int i = tid; i < 64 * 64; i += 128) {
            const int k = i >> 6, c = i & 63;
            const int kk = k >> 5, kc = k & 31;
            const int hf = (kc >> 3) & 1, pos = (kc & 7) + 8 * (kc >> 4);
            Bf[c >> 4][kk][(c & 15) + 16 * hf][pos] =
                (_Float16)Bm[(size_t)(k0 + k) * Ncol + n0 + c];
        }
        __syncthreads();

#pragma unroll
        for (int kk = 0; kk < 2; ++kk) {
            const v16h a = *(const v16h*)Af[wave][kk][lane];
#pragma unroll
            for (int nt = 0; nt < 4; ++nt) {
                const v16h b = *(const v16h*)Bf[nt][kk][lane];
                acc[nt] = __builtin_amdgcn_wmma_f32_16x16x32_f16(
                    false, a, false, b, (short)0, acc[nt], false, false);
            }
        }
        __syncthreads();
    }

    // D layout: VGPR r -> row r + 8*(lane>=16), col = lane&15
    const int half = lane >> 4;
    const int rcL  = lane & 15;
#pragma unroll
    for (int nt = 0; nt < 4; ++nt) {
        const int j = n0 + nt * 16 + rcL;            // output column
        if (MODE == OUT_F32_ROWMAJOR) {
            float* C = (float*)Cv;
#pragma unroll
            for (int r = 0; r < 8; ++r)
                C[(size_t)(m0 + wave * 16 + r + 8 * half) * Ncol + j] = acc[nt][r];
        } else if (MODE == OUT_F16_BNSH) {           // [B,N,S,H] f16
            _Float16* C = (_Float16*)Cv;
            const int n = j >> 6, h = j & 63;
#pragma unroll
            for (int r = 0; r < 8; ++r) {
                const int m = m0 + wave * 16 + r + 8 * half;
                const int b = m >> 11, s = m & 2047;
                C[((size_t)(b * Nn + n) * Ss + s) * Hh + h] = (_Float16)acc[nt][r];
            }
        } else {                                     // OUT_F16_BNHS: [B,N,H,S] f16
            const int n = j >> 6, h = j & 63;
            const int s0 = m0 + wave * 16 + 8 * half;  // 8 consecutive s
            const int b = s0 >> 11;
            v8h hv;
#pragma unroll
            for (int r = 0; r < 8; ++r) hv[r] = (_Float16)acc[nt][r];
            *(v8h*)&((_Float16*)Cv)[((size_t)(b * Nn + n) * Hh + h) * Ss + (s0 & 2047)] = hv;
        }
    }
}

// ---------------------------------------------------------------------------
// Fused attention per (b, n, 16-query-row tile).
// Qh,Kh: f16 [B,N,S,H]; Vt: f16 [B,N,H,S]; attn: f32 [B,N,S,S]; ctx: f16 [B,S,N*H]
// Dynamic LDS: logits 16x2048 f32 (128KB, fits CDNA5 320KB WGP LDS).
// attn tile streamed LDS->HBM with GLOBAL_STORE_ASYNC_FROM_LDS_B128,
// overlapped with the Phase-3 WMMA loop (ASYNCcnt drained at kernel end).
// ---------------------------------------------------------------------------
__global__ __launch_bounds__(128) void attn_fused(
    const _Float16* __restrict__ Qh, const _Float16* __restrict__ Kh,
    const _Float16* __restrict__ Vt, float* __restrict__ attn,
    _Float16* __restrict__ ctx)
{
    extern __shared__ char smem[];
    float* lg = (float*)smem;                            // [16][2048]
    __shared__ alignas(32) _Float16 qsf[2][32][16];      // Q tile, frag-major, 2 k-halves
    __shared__ float red[16][8];

    const int tid  = threadIdx.x;
    const int wave = tid >> 5;
    const int lane = tid & 31;
    const int half = lane >> 4;
    const int rc   = lane & 15;

    const int bid   = blockIdx.x;
    const int ftile = bid % FTILES;
    const int n     = (bid / FTILES) % Nn;
    const int b     = bid / (FTILES * Nn);
    const int f0    = ftile * 16;

    const size_t headQK = (size_t)(b * Nn + n) * Ss * Hh;   // base of [S,H] slab
    const size_t headV  = (size_t)(b * Nn + n) * Hh * Ss;   // base of [H,S] slab

    // stage Q tile (16x64 f16) into fragment-major layout
    for (int i = tid; i < 16 * 64; i += 128) {
        const int r = i >> 6, h = i & 63;
        const int c = h & 31, kk = h >> 5;
        const int hf = (c >> 3) & 1, pos = (c & 7) + 8 * (c >> 4);
        qsf[kk][r + 16 * hf][pos] = Qh[headQK + (size_t)(f0 + r) * Hh + h];
    }
    __syncthreads();

    // ---- Phase 1: logits = (Q K^T)/8 + causal mask ----
    for (int tt = wave; tt < FTILES; tt += 4) {
        const int t0 = tt * 16;
        if (tt + 4 < FTILES)
            __builtin_prefetch(&Kh[headQK + (size_t)((tt + 4) * 16 + rc) * Hh], 0, 0);

        const _Float16* krow = &Kh[headQK + (size_t)(t0 + rc) * Hh];
        v8f acc = {};
#pragma unroll
        for (int kk = 0; kk < 2; ++kk) {
            const v16h a  = *(const v16h*)qsf[kk][lane];
            const v8h  lo = *(const v8h*)(krow + kk * 32 + 8 * half);
            const v8h  hi = *(const v8h*)(krow + kk * 32 + 16 + 8 * half);
            const v16h bf = __builtin_shufflevector(lo, hi,
                0, 1, 2, 3, 4, 5, 6, 7, 8, 9, 10, 11, 12, 13, 14, 15);
            acc = __builtin_amdgcn_wmma_f32_16x16x32_f16(
                false, a, false, bf, (short)0, acc, false, false);
        }
#pragma unroll
        for (int r = 0; r < 8; ++r) {
            const int row = r + 8 * half;
            const int f = f0 + row, t = t0 + rc;
            lg[row * 2048 + t] = acc[r] * 0.125f + (t > f ? -1e9f : 0.0f);
        }
    }
    __syncthreads();

    // ---- Phase 2: row softmax, then async LDS->global attn stream ----
    {
        const int row = tid >> 3, j = tid & 7;
        float pm = -3.4e38f;
        for (int t = j; t < 2048; t += 8) pm = fmaxf(pm, lg[row * 2048 + t]);
        red[row][j] = pm;
        __syncthreads();
        if (tid < 16) {
            float m = red[tid][0];
            for (int q = 1; q < 8; ++q) m = fmaxf(m, red[tid][q]);
            red[tid][0] = m;
        }
        __syncthreads();
        const float m = red[row][0];
        float ps = 0.f;
        for (int t = j; t < 2048; t += 8) {
            float e = __expf(lg[row * 2048 + t] - m);
            lg[row * 2048 + t] = e;
            ps += e;
        }
        __syncthreads();
        red[row][j] = ps;
        __syncthreads();
        if (tid < 16) {
            float s = 0.f;
            for (int q = 0; q < 8; ++q) s += red[tid][q];
            red[tid][0] = 1.0f / s;
        }
        __syncthreads();
        const float inv = red[row][0];
        for (int t = j; t < 2048; t += 8) lg[row * 2048 + t] *= inv;
        __syncthreads();

        const size_t tbase = ((size_t)(b * Nn + n) * Ss + f0) * Ss;
#if defined(HAVE_ASYNC_STORE)
        // 16B per lane per issue; tile is contiguous in both LDS and global.
        for (int i = tid * 4; i < 16 * 2048; i += 128 * 4)
            __builtin_amdgcn_global_store_async_from_lds_b128(
                (g4i_t)(attn + tbase + i), (l4i_t)(lg + i), 0, 0);
        // no wait here: stores only READ lg; Phase 3 also only reads lg.
#else
        for (int i = tid; i < 16 * 2048; i += 128) attn[tbase + i] = lg[i];
#endif
        __syncthreads();
    }

    // ---- Phase 3: ctx[16,64] = attn @ V ; one 16-col tile per wave ----
    {
        const int h0 = wave * 16;
        const float*    arow = lg + rc * 2048;
        const _Float16* vrow = &Vt[headV + (size_t)(h0 + rc) * Ss];
        v8f acc = {};
        for (int t0 = 0; t0 < 2048; t0 += 32) {
            const v8f a0 = *(const v8f*)(arow + t0 + 8 * half);        // 32B aligned
            const v8f a1 = *(const v8f*)(arow + t0 + 16 + 8 * half);
            v16h a;
#pragma unroll
            for (int i = 0; i < 8; ++i) { a[i] = (_Float16)a0[i]; a[i + 8] = (_Float16)a1[i]; }
            const v8h lo = *(const v8h*)(vrow + t0 + 8 * half);
            const v8h hi = *(const v8h*)(vrow + t0 + 16 + 8 * half);
            const v16h bf = __builtin_shufflevector(lo, hi,
                0, 1, 2, 3, 4, 5, 6, 7, 8, 9, 10, 11, 12, 13, 14, 15);
            acc = __builtin_amdgcn_wmma_f32_16x16x32_f16(
                false, a, false, bf, (short)0, acc, false, false);
        }
#pragma unroll
        for (int r = 0; r < 8; ++r)
            ctx[(size_t)(b * Ss + f0 + r + 8 * half) * Dd + n * Hh + h0 + rc] =
                (_Float16)acc[r];
    }

#if defined(HAVE_ASYNC_STORE) && defined(HAVE_WAIT_ASYNC)
    __builtin_amdgcn_s_wait_asynccnt(0);   // drain attn stream (ENDPGM also waits)
#endif
}

// ---------------------------------------------------------------------------
extern "C" void kernel_launch(void* const* d_in, const int* in_sizes, int n_in,
                              void* d_out, int out_size, void* d_ws, size_t ws_size,
                              hipStream_t stream) {
    (void)in_sizes; (void)n_in; (void)out_size; (void)ws_size;
    const float* v_in = (const float*)d_in[0];
    const float* k_in = (const float*)d_in[1];
    const float* q_in = (const float*)d_in[2];
    // d_in[3] = causal mask, applied analytically in-kernel
    const float* Wq = (const float*)d_in[4];   // [D, N*H] row-major
    const float* Wk = (const float*)d_in[5];
    const float* Wv = (const float*)d_in[6];
    const float* Wo = (const float*)d_in[7];   // [N*H, D] row-major

    float* out  = (float*)d_out;                          // [B,S,D] f32
    float* attn = out + (size_t)Bb * Ss * Dd;             // [B,N,S,S] f32

    _Float16* Qh = (_Float16*)d_ws;                       // [B,N,S,H] f16 (8MB)
    _Float16* Kh = Qh + (size_t)MROWS * Dd;
    _Float16* Vt = Kh + (size_t)MROWS * Dd;               // [B,N,H,S] f16
    _Float16* Cx = Vt + (size_t)MROWS * Dd;               // ctx [B,S,N*H] f16

    dim3 gg(Dd / 64, MROWS / 64), bb(128);
    wmma_gemm<float, OUT_F16_BNSH><<<gg, bb, 0, stream>>>(q_in, Wq, Qh, MROWS, Dd, Dd);
    wmma_gemm<float, OUT_F16_BNSH><<<gg, bb, 0, stream>>>(k_in, Wk, Kh, MROWS, Dd, Dd);
    wmma_gemm<float, OUT_F16_BNHS><<<gg, bb, 0, stream>>>(v_in, Wv, Vt, MROWS, Dd, Dd);

    const size_t shmem = 16 * 2048 * sizeof(float);       // logits tile
    attn_fused<<<Bb * Nn * FTILES, 128, shmem, stream>>>(Qh, Kh, Vt, attn, Cx);

    wmma_gemm<_Float16, OUT_F32_ROWMAJOR><<<gg, bb, 0, stream>>>(Cx, Wo, out, MROWS, Dd, Dd);
}
